// RNNDecoder_68478958567972
// MI455X (gfx1250) — matrix operand order repaired and checked by reading, not compiled, not run on hardware
//
#include <hip/hip_runtime.h>
#include <hip/hip_bf16.h>
#include <math.h>

typedef __bf16 bf16_t;
typedef __attribute__((ext_vector_type(16))) __bf16 v16bf;
typedef __attribute__((ext_vector_type(8)))  __bf16 v8bf;
typedef __attribute__((ext_vector_type(8)))  float   v8f;
typedef __attribute__((ext_vector_type(4)))  unsigned int v4u;
typedef __attribute__((ext_vector_type(4)))  int v4i;
typedef __attribute__((ext_vector_type(8)))  int v8i;

#define DEV __device__ __forceinline__

namespace cfg {
constexpr int S  = 48;
constexpr int B  = 32;
constexpr int SC = 64;
constexpr int E  = 512;
constexpr int C  = 512;
constexpr int H  = 512;
constexpr int V  = 50257;
constexpr int R  = S * B;          // 1536 (s major, b minor)
constexpr int G3 = 3 * H;          // 1536
constexpr int XK = H + C;          // 1024

// fused logsumexp pass over vocab
constexpr int LSE_WAVES  = 8;
constexpr int LSE_TPW    = 16;                         // 16x16 col tiles per wave
constexpr int LSE_CHUNK  = LSE_WAVES * LSE_TPW * 16;   // 2048 cols per block
constexpr int LSE_NCHUNK = (V + LSE_CHUNK - 1) / LSE_CHUNK;  // 25
constexpr int NPART      = LSE_NCHUNK * LSE_WAVES;     // 200 partials per row

// cooperative recurrence grid
constexpr int RWG = 12;
constexpr int RTH = 256;
}

// ---------------- workspace layout (bytes) ----------------
constexpr size_t AL(size_t x) { return (x + 255) & ~size_t(255); }
constexpr size_t OFF_GIX    = 0;                                                  // R x 3H f32
constexpr size_t OFF_HALL   = AL(OFF_GIX    + sizeof(float) * cfg::R * cfg::G3);  // R x H f32
constexpr size_t OFF_CALL   = AL(OFF_HALL   + sizeof(float) * cfg::R * cfg::H);   // R x C f32
constexpr size_t OFF_Q      = AL(OFF_CALL   + sizeof(float) * cfg::R * cfg::C);   // B x C f32
constexpr size_t OFF_SCO    = AL(OFF_Q      + sizeof(float) * cfg::B * cfg::C);   // SC x B
constexpr size_t OFF_ALP    = AL(OFF_SCO    + sizeof(float) * cfg::SC * cfg::B);
constexpr size_t OFF_MEAN   = AL(OFF_ALP    + sizeof(float) * cfg::SC * cfg::B);  // B x C
constexpr size_t OFF_HCUR   = AL(OFF_MEAN   + sizeof(float) * cfg::B * cfg::C);   // B x H f32
constexpr size_t OFF_HBFA   = AL(OFF_HCUR   + sizeof(float) * cfg::B * cfg::H);   // bf16
constexpr size_t OFF_HBFB   = AL(OFF_HBFA   + 2ull * cfg::B * cfg::H);
constexpr size_t OFF_CTXBF  = AL(OFF_HBFB   + 2ull * cfg::B * cfg::H);
constexpr size_t OFF_XBF    = AL(OFF_CTXBF  + 2ull * cfg::B * cfg::C);            // R x XK bf16
constexpr size_t OFF_HBFALL = AL(OFF_XBF    + 2ull * cfg::R * cfg::XK);           // R x H bf16
constexpr size_t OFF_PMAXC  = AL(OFF_HBFALL + 2ull * cfg::R * cfg::H);
constexpr size_t OFF_PSUMC  = AL(OFF_PMAXC  + sizeof(float) * cfg::R * cfg::NPART);
constexpr size_t OFF_PMAXB  = AL(OFF_PSUMC  + sizeof(float) * cfg::R * cfg::NPART);
constexpr size_t OFF_PSUMB  = AL(OFF_PMAXB  + sizeof(float) * cfg::R * cfg::NPART);
constexpr size_t OFF_CETGT  = AL(OFF_PSUMB  + sizeof(float) * cfg::R * cfg::NPART);
constexpr size_t OFF_BOWTGT = AL(OFF_CETGT  + sizeof(float) * cfg::R);
constexpr size_t OFF_BAR    = AL(OFF_BOWTGT + sizeof(float) * cfg::R * 5);

// ---------------- WMMA helpers (CDNA5 bf16 16x16x32 layouts) ----------------
DEV v8f zero8() {
  v8f z;
#pragma unroll
  for (int i = 0; i < 8; ++i) z[i] = 0.f;
  return z;
}

DEV v8f wmma_bf16(v16bf a, v16bf b, v8f c) {
  // D = A(16x32) * B(32x16) + C, f32 accumulate
  return __builtin_amdgcn_wmma_f32_16x16x32_bf16(false, a, false, b, (short)0, c,
                                                 false, false);
}

// A fragment (16x32) from row-major f32 [.. x lda] with on-the-fly bf16 convert.
// lane<16: M=lane, K in {0..7, 16..23}; lane>=16: M=lane-16, K in {8..15, 24..31}
DEV v16bf a_frag_f32(const float* A, int lda, int row0, int k0) {
  int lane = threadIdx.x & 31;
  const float* p = A + (size_t)(row0 + (lane & 15)) * lda + k0 + ((lane & 16) ? 8 : 0);
  v16bf a;
#pragma unroll
  for (int j = 0; j < 8; ++j) { a[j] = (__bf16)p[j]; a[8 + j] = (__bf16)p[16 + j]; }
  return a;
}

// A fragment from row-major bf16 (global or LDS) -- two 16B vector loads per lane
DEV v16bf a_frag_bf16(const bf16_t* A, int lda, int row0, int k0) {
  int lane = threadIdx.x & 31;
  const bf16_t* p = A + (size_t)(row0 + (lane & 15)) * lda + k0 + ((lane & 16) ? 8 : 0);
  v8bf lo = *(const v8bf*)p;
  v8bf hi = *(const v8bf*)(p + 16);
  v16bf a;
#pragma unroll
  for (int j = 0; j < 8; ++j) { a[j] = lo[j]; a[8 + j] = hi[j]; }
  return a;
}

// B fragment (32x16) from row-major f32 [K x ldb]; n = this lane's absolute column.
// lane<16: N=lane, K in {0..7, 16..23}; lane>=16: N=lane-16, K in {8..15, 24..31}
DEV v16bf b_frag_f32(const float* Bm, int ldb, int k0, int n) {
  int lane = threadIdx.x & 31;
  const float* p = Bm + (size_t)(k0 + ((lane & 16) ? 8 : 0)) * ldb + n;
  v16bf b;
#pragma unroll
  for (int j = 0; j < 16; ++j) b[j] = (__bf16)p[(size_t)((j < 8) ? j : (j + 8)) * ldb];
  return b;
}

// ---------------- small prep kernels ----------------
__global__ void k_init(unsigned* bar) { bar[0] = 0u; bar[1] = 0u; }

__global__ void k_mean_ctx(const float* ctx, float* mean) {
  int t = blockIdx.x * blockDim.x + threadIdx.x;
  if (t >= cfg::B * cfg::C) return;
  int b = t >> 9, c = t & 511;
  float s = 0.f;
  for (int sc = 0; sc < cfg::SC; ++sc)
    s += ctx[((size_t)sc * cfg::B + b) * cfg::C + c];
  mean[b * cfg::C + c] = s * (1.f / cfg::SC);
}

// h0 = mean(context) @ W_c2h + b_c2h   [32 x 512], 64 wave-tiles
__global__ __launch_bounds__(256) void k_h0(const float* mean, const float* Wc2h,
                                            const float* bc2h, float* hcur, bf16_t* hbf) {
  int gw = blockIdx.x * 8 + (threadIdx.x >> 5);
  int lane = threadIdx.x & 31;
  if (gw >= 64) return;
  int mt = gw >> 5, nt = gw & 31;
  int n = nt * 16 + (lane & 15);
  v8f acc = zero8();
  for (int k = 0; k < cfg::H; k += 32) {
    v16bf a = a_frag_f32(mean, cfg::C, mt * 16, k);
    v16bf b = b_frag_f32(Wc2h, cfg::H, k, n);
    acc = wmma_bf16(a, b, acc);
  }
  int mb = mt * 16 + ((lane & 16) ? 8 : 0);
  float bias = bc2h[n];
#pragma unroll
  for (int r = 0; r < 8; ++r) {
    float v = acc[r] + bias;
    hcur[(mb + r) * cfg::H + n] = v;
    hbf[(mb + r) * cfg::H + n] = (bf16_t)v;
  }
}

// gi_x = emb @ W_ih[0:E,:] + b_ih  (no recurrence dependency -> fully parallel)
__global__ __launch_bounds__(256) void k_gix(const float* emb, const float* Wih,
                                             const float* bih, float* gix) {
  int bx = blockIdx.x;
  int rt = bx / 12, cc = bx % 12;
  int wid = threadIdx.x >> 5, lane = threadIdx.x & 31;
  int n = cc * 128 + wid * 16 + (lane & 15);
  v8f acc = zero8();
  for (int k = 0; k < cfg::E; k += 32) {
    v16bf a = a_frag_f32(emb, cfg::E, rt * 16, k);
    v16bf b = b_frag_f32(Wih, cfg::G3, k, n);
    acc = wmma_bf16(a, b, acc);
  }
  int mb = rt * 16 + ((lane & 16) ? 8 : 0);
  float bias = bih[n];
#pragma unroll
  for (int r = 0; r < 8; ++r)
    gix[(size_t)(mb + r) * cfg::G3 + n] = acc[r] + bias;
}

// ---------------- cooperative recurrence ----------------
DEV void grid_barrier(unsigned* cnt, unsigned* gen) {
  __threadfence();
  __syncthreads();
  if (threadIdx.x == 0) {
    unsigned g = __atomic_load_n(gen, __ATOMIC_ACQUIRE);
    if (atomicAdd(cnt, 1u) == (unsigned)cfg::RWG - 1u) {
      *cnt = 0u;
      __threadfence();
      __atomic_store_n(gen, g + 1u, __ATOMIC_RELEASE);
    } else {
      while (__atomic_load_n(gen, __ATOMIC_ACQUIRE) == g)
        __builtin_amdgcn_s_sleep(1);
    }
  }
  __syncthreads();
}

__global__ __launch_bounds__(cfg::RTH) void k_recurrence(
    const float* context, const float* Wattn, const float* Wih, const float* Whh,
    const float* bhh, const float* gix, float* q, float* sco, float* alp,
    float* hcur, bf16_t* hbfA, bf16_t* hbfB, bf16_t* ctxbf,
    float* hall, float* ctx_all, unsigned* bar) {
  const int tid = threadIdx.x;
  const int lane = tid & 31;
  const int gw = blockIdx.x * (cfg::RTH / 32) + (tid >> 5);  // 0..95
  const int gtid = blockIdx.x * cfg::RTH + tid;              // 0..3071
  unsigned* cnt = bar;
  unsigned* gen = bar + 1;

  for (int s = 0; s < cfg::S; ++s) {
    const bf16_t* hbr = (s & 1) ? hbfB : hbfA;  // h(t) bf16 read copy
    bf16_t* hbw = (s & 1) ? hbfA : hbfB;        // h(t+1) write copy

    // ---- phase A: q = h @ W_attn  [32 x 512] (64 wave-tiles) ----
    if (gw < 64) {
      int mt = gw >> 5, nt = gw & 31;
      int n = nt * 16 + (lane & 15);
      v8f acc = zero8();
      for (int k = 0; k < cfg::H; k += 32) {
        v16bf a = a_frag_bf16(hbr, cfg::H, mt * 16, k);
        v16bf b = b_frag_f32(Wattn, cfg::C, k, n);
        acc = wmma_bf16(a, b, acc);
      }
      int mb = mt * 16 + ((lane & 16) ? 8 : 0);
#pragma unroll
      for (int r = 0; r < 8; ++r) q[(mb + r) * cfg::C + n] = acc[r];
    }
    grid_barrier(cnt, gen);

    // ---- phase B1: scores[sc][b] = context[sc,b,:] . q[b,:] ----
    if (gtid < cfg::SC * cfg::B) {
      int b = gtid & 31, sc = gtid >> 5;
      const float* cp = context + ((size_t)sc * cfg::B + b) * cfg::C;
      const float* qp = q + b * cfg::C;
      float d = 0.f;
#pragma unroll 4
      for (int k = 0; k < cfg::C; ++k) d += cp[k] * qp[k];
      sco[sc * cfg::B + b] = d;
    }
    grid_barrier(cnt, gen);

    // ---- phase B2: softmax over sc per batch column ----
    if (gtid < cfg::B) {
      int b = gtid;
      float m = -INFINITY;
      for (int sc = 0; sc < cfg::SC; ++sc) m = fmaxf(m, sco[sc * cfg::B + b]);
      float ssum = 0.f;
      for (int sc = 0; sc < cfg::SC; ++sc) ssum += expf(sco[sc * cfg::B + b] - m);
      float inv = 1.f / ssum;
      for (int sc = 0; sc < cfg::SC; ++sc)
        alp[sc * cfg::B + b] = expf(sco[sc * cfg::B + b] - m) * inv;
    }
    grid_barrier(cnt, gen);

    // ---- phase B3: ctx[b][c] = sum_sc alpha * context ----
    for (int t = gtid; t < cfg::B * cfg::C; t += cfg::RWG * cfg::RTH) {
      int b = t >> 9, c = t & 511;
      float acc = 0.f;
      for (int sc = 0; sc < cfg::SC; ++sc)
        acc += alp[sc * cfg::B + b] * context[((size_t)sc * cfg::B + b) * cfg::C + c];
      ctx_all[((size_t)(s * cfg::B + b)) * cfg::C + c] = acc;
      ctxbf[b * cfg::C + c] = (bf16_t)acc;
    }
    grid_barrier(cnt, gen);

    // ---- phase C: GRU gates + h update (64 wave-tiles; r,z,n share column c) ----
    if (gw < 64) {
      int mt = gw >> 5, ct = gw & 31;
      int n = ct * 16 + (lane & 15);  // column within H
      v8f ir = zero8(), iz = zero8(), in_ = zero8();
      v8f hr = zero8(), hz = zero8(), hn = zero8();
      const float* WihC = Wih + (size_t)cfg::E * cfg::G3;  // ctx half of W_ih
      for (int k = 0; k < cfg::C; k += 32) {
        v16bf ac = a_frag_bf16(ctxbf, cfg::C, mt * 16, k);
        ir  = wmma_bf16(ac, b_frag_f32(WihC, cfg::G3, k, n), ir);
        iz  = wmma_bf16(ac, b_frag_f32(WihC, cfg::G3, k, 512 + n), iz);
        in_ = wmma_bf16(ac, b_frag_f32(WihC, cfg::G3, k, 1024 + n), in_);
        v16bf ah = a_frag_bf16(hbr, cfg::H, mt * 16, k);
        hr  = wmma_bf16(ah, b_frag_f32(Whh, cfg::G3, k, n), hr);
        hz  = wmma_bf16(ah, b_frag_f32(Whh, cfg::G3, k, 512 + n), hz);
        hn  = wmma_bf16(ah, b_frag_f32(Whh, cfg::G3, k, 1024 + n), hn);
      }
      int mb = mt * 16 + ((lane & 16) ? 8 : 0);
      float br = bhh[n], bz = bhh[512 + n], bn = bhh[1024 + n];
#pragma unroll
      for (int r = 0; r < 8; ++r) {
        int row = mb + r;  // batch index 0..31
        const float* gp = gix + ((size_t)(s * cfg::B + row)) * cfg::G3;
        float vir = ir[r] + gp[n];
        float viz = iz[r] + gp[512 + n];
        float vin = in_[r] + gp[1024 + n];
        float rg = 1.f / (1.f + expf(-(vir + hr[r] + br)));
        float zg = 1.f / (1.f + expf(-(viz + hz[r] + bz)));
        float nn = tanhf(vin + rg * (hn[r] + bn));
        float hold = hcur[row * cfg::H + n];
        float hnew = (1.f - zg) * nn + zg * hold;
        hall[((size_t)(s * cfg::B + row)) * cfg::H + n] = hnew;
        hcur[row * cfg::H + n] = hnew;
        hbw[row * cfg::H + n] = (bf16_t)hnew;
      }
    }
    grid_barrier(cnt, gen);
  }
}

// pack [h;ctx] rows to bf16 operand matrices for the vocab passes
__global__ void k_pack(const float* hall, const float* ctx_all, bf16_t* xbf,
                       bf16_t* hbfall) {
  int t = blockIdx.x * blockDim.x + threadIdx.x;
  if (t >= cfg::R * cfg::XK) return;
  int r = t >> 10, c = t & 1023;
  float v = (c < 512) ? hall[(size_t)r * 512 + c] : ctx_all[(size_t)r * 512 + (c - 512)];
  xbf[(size_t)r * cfg::XK + c] = (bf16_t)v;
  if (c < 512) hbfall[(size_t)r * 512 + c] = (bf16_t)v;
}

// ---------------- fused vocab matmul -> online logsumexp partials ----------------
template <int KD>
__global__ __launch_bounds__(256) void k_lse_partials(const bf16_t* Abf, const float* W,
                                                      const float* bias, float* pmax,
                                                      float* psum) {
  __shared__ __align__(16) bf16_t sA[16 * KD];  // 16-row slab of A, bf16
  const int rt = blockIdx.x / cfg::LSE_NCHUNK;
  const int ch = blockIdx.x % cfg::LSE_NCHUNK;
  const int wid = threadIdx.x >> 5, lane = threadIdx.x & 31;

  // ---- stage A slab (16 rows x KD bf16, contiguous in memory) into LDS ----
#if __has_builtin(__builtin_amdgcn_tensor_load_to_lds) && \
    __has_builtin(__builtin_amdgcn_s_wait_tensorcnt)
  if (threadIdx.x < 32) {  // one wave issues the Tensor Data Mover DMA
    // D# descriptor, 2D tensor [R x KD] bf16, tile = 16 rows x KD cols.
    unsigned long long ga =
        (unsigned long long)(size_t)(Abf + (size_t)rt * 16 * KD);
    v4u g0;
    g0[0] = 1u;                                  // count=1, user descriptor
    g0[1] = 0u;                                  // lds_addr: sA is the only LDS object
    g0[2] = (unsigned)(ga & 0xFFFFFFFFu);
    g0[3] = (unsigned)((ga >> 32) & 0x01FFFFFFu) | (2u << 30);  // type=2 (image)
    v8i g1;
    g1[0] = (int)(1u << 16);                                    // data_size=1 -> 2B
    g1[1] = (int)(((unsigned)KD & 0xFFFFu) << 16);              // tensor_dim0 lo16
    g1[2] = (int)(((unsigned)KD >> 16) |
                  (((unsigned)cfg::R & 0xFFFFu) << 16));        // td0 hi | td1 lo
    g1[3] = (int)(((unsigned)cfg::R >> 16) |
                  (((unsigned)KD & 0xFFFFu) << 16));            // td1 hi | tile_dim0
    g1[4] = 16;                                                 // tile_dim1=16 rows
    g1[5] = KD;                                                 // tensor_dim0_stride
    g1[6] = 0;
    g1[7] = 0;
    v4i z4 = {0, 0, 0, 0};
#if defined(__clang_major__) && (__clang_major__ >= 23)
    v8i z8 = {0, 0, 0, 0, 0, 0, 0, 0};
    __builtin_amdgcn_tensor_load_to_lds(g0, g1, z4, z4, z8, 0);
#else
    __builtin_amdgcn_tensor_load_to_lds(g0, g1, z4, z4, 0);
#endif
    __builtin_amdgcn_s_wait_tensorcnt(0);
  }
  __syncthreads();
#else
  {
    const uint4* src = (const uint4*)(Abf + (size_t)rt * 16 * KD);
    uint4* dst = (uint4*)sA;
    for (int i = threadIdx.x; i < 2 * KD; i += 256) dst[i] = src[i];
  }
  __syncthreads();
#endif

  float rm[8], rs[8];
#pragma unroll
  for (int r = 0; r < 8; ++r) { rm[r] = -INFINITY; rs[r] = 0.f; }

  const int colw = ch * cfg::LSE_CHUNK + wid * (cfg::LSE_TPW * 16);
  for (int t = 0; t < cfg::LSE_TPW; ++t) {
    int col0 = colw + t * 16;
    if (col0 >= cfg::V) break;
    int n = col0 + (lane & 15);
    bool ok = (n < cfg::V);
    int ncl = ok ? n : (cfg::V - 1);
    v8f acc = zero8();
    for (int k = 0; k < KD; k += 32) {
      if (k + 32 < KD)  // stream next weight rows toward L2
        __builtin_prefetch(W + (size_t)(k + 32) * cfg::V + ncl, 0, 1);
      v16bf a = a_frag_bf16(sA, KD, 0, k);
      v16bf b = b_frag_f32(W, cfg::V, k, ncl);
      acc = wmma_bf16(a, b, acc);
    }
    float bv = bias[ncl];
#pragma unroll
    for (int r = 0; r < 8; ++r) {  // per-row reduce over the 16 lanes of this half
      float v = ok ? (acc[r] + bv) : -INFINITY;
      float tm = v;
#pragma unroll
      for (int m = 1; m < 16; m <<= 1) tm = fmaxf(tm, __shfl_xor(tm, m, 16));
      float e = (tm == -INFINITY) ? 0.f : expf(v - tm);
      float ts = e;
#pragma unroll
      for (int m = 1; m < 16; m <<= 1) ts += __shfl_xor(ts, m, 16);
      if (tm != -INFINITY) {  // online merge into running (max, sumexp)
        float nm = fmaxf(rm[r], tm);
        rs[r] = rs[r] * expf(rm[r] - nm) + ts * expf(tm - nm);
        rm[r] = nm;
      }
    }
  }
  if ((lane & 15) == 0) {
    int pidx = ch * cfg::LSE_WAVES + wid;
    int rbase = rt * 16 + ((lane & 16) ? 8 : 0);
#pragma unroll
    for (int r = 0; r < 8; ++r) {
      pmax[(size_t)(rbase + r) * cfg::NPART + pidx] = rm[r];
      psum[(size_t)(rbase + r) * cfg::NPART + pidx] = rs[r];
    }
  }
}

// gather logits actually needed by the losses (1536 x 6 dot products, f32)
__global__ __launch_bounds__(256) void k_tgt(const float* hall, const float* ctx_all,
                                             const float* Wout, const float* bout,
                                             const float* Wbow, const float* bbow,
                                             const int* target, float* cetgt,
                                             float* bowtgt) {
  int row = blockIdx.x * 8 + (threadIdx.x >> 5);
  int lane = threadIdx.x & 31;
  if (row >= cfg::R) return;
  int s = row >> 5, b = row & 31;
  int tg = target[row];
  float acc = 0.f;
  for (int k = lane; k < cfg::XK; k += 32) {
    float x = (k < 512) ? hall[(size_t)row * 512 + k]
                        : ctx_all[(size_t)row * 512 + (k - 512)];
    acc += x * Wout[(size_t)k * cfg::V + tg];
  }
#pragma unroll
  for (int m = 16; m >= 1; m >>= 1) acc += __shfl_xor(acc, m, 32);
  if (lane == 0) cetgt[row] = acc + bout[tg];
#pragma unroll
  for (int j = 0; j < 5; ++j) {
    int sj = s + j;
    float d = 0.f;
    int tb = 0;
    if (sj < cfg::S) {
      tb = target[sj * cfg::B + b];
      for (int k = lane; k < cfg::H; k += 32)
        d += hall[(size_t)row * 512 + k] * Wbow[(size_t)k * cfg::V + tb];
    }
#pragma unroll
    for (int m = 16; m >= 1; m >>= 1) d += __shfl_xor(d, m, 32);
    if (lane == 0) bowtgt[row * 5 + j] = (sj < cfg::S) ? (d + bbow[tb]) : 0.f;
  }
}

__global__ void k_finalize(const float* pmaxc, const float* psumc, const float* pmaxb,
                           const float* psumb, const float* cetgt, const float* bowtgt,
                           const float* mask, float* out) {
  __shared__ float sce[256], sbw[256];
  float ce = 0.f, bw = 0.f;
  for (int row = threadIdx.x; row < cfg::R; row += 256) {
    float m = -INFINITY, ss = 0.f;
    for (int p = 0; p < cfg::NPART; ++p) {
      float pm = pmaxc[(size_t)row * cfg::NPART + p];
      if (pm == -INFINITY) continue;
      float ps = psumc[(size_t)row * cfg::NPART + p];
      if (pm > m) { ss = ss * expf(m - pm) + ps; m = pm; }
      else        { ss += ps * expf(pm - m); }
    }
    float lse = m + logf(ss);
    ce += (lse - cetgt[row]) * mask[row];

    float mb = -INFINITY, sb = 0.f;
    for (int p = 0; p < cfg::NPART; ++p) {
      float pm = pmaxb[(size_t)row * cfg::NPART + p];
      if (pm == -INFINITY) continue;
      float ps = psumb[(size_t)row * cfg::NPART + p];
      if (pm > mb) { sb = sb * expf(mb - pm) + ps; mb = pm; }
      else         { sb += ps * expf(pm - mb); }
    }
    float lseb = mb + logf(sb);
    int s = row >> 5;
    int wv = cfg::S - s; if (wv > 5) wv = 5;
    float inv = 1.f / (float)(cfg::S * wv * cfg::B);
    for (int j = 0; j < wv; ++j) bw += (lseb - bowtgt[row * 5 + j]) * inv;
  }
  sce[threadIdx.x] = ce; sbw[threadIdx.x] = bw;
  __syncthreads();
  for (int st = 128; st > 0; st >>= 1) {
    if (threadIdx.x < st) {
      sce[threadIdx.x] += sce[threadIdx.x + st];
      sbw[threadIdx.x] += sbw[threadIdx.x + st];
    }
    __syncthreads();
  }
  if (threadIdx.x == 0) {
    out[0] = sce[0] / (float)(cfg::S * cfg::B);
    out[1] = sbw[0];
  }
}

// ---------------- host launch ----------------
extern "C" void kernel_launch(void* const* d_in, const int* in_sizes, int n_in,
                              void* d_out, int out_size, void* d_ws, size_t ws_size,
                              hipStream_t stream) {
  (void)in_sizes; (void)n_in; (void)out_size; (void)ws_size;
  const float* emb   = (const float*)d_in[0];
  const float* ctx   = (const float*)d_in[1];
  const float* mask  = (const float*)d_in[2];
  const float* Wattn = (const float*)d_in[3];
  const float* Wc2h  = (const float*)d_in[4];
  const float* bc2h  = (const float*)d_in[5];
  const float* Wih   = (const float*)d_in[6];
  const float* Whh   = (const float*)d_in[7];
  const float* bih   = (const float*)d_in[8];
  const float* bhh   = (const float*)d_in[9];
  const float* Wout  = (const float*)d_in[10];
  const float* bout  = (const float*)d_in[11];
  const float* Wbow  = (const float*)d_in[12];
  const float* bbow  = (const float*)d_in[13];
  const int*   tgt   = (const int*)d_in[14];
  float* out = (float*)d_out;

  char* w = (char*)d_ws;
  float*  gix    = (float*)(w + OFF_GIX);
  float*  hall   = (float*)(w + OFF_HALL);
  float*  call_  = (float*)(w + OFF_CALL);
  float*  q      = (float*)(w + OFF_Q);
  float*  sco    = (float*)(w + OFF_SCO);
  float*  alp    = (float*)(w + OFF_ALP);
  float*  mean   = (float*)(w + OFF_MEAN);
  float*  hcur   = (float*)(w + OFF_HCUR);
  bf16_t* hbfA   = (bf16_t*)(w + OFF_HBFA);
  bf16_t* hbfB   = (bf16_t*)(w + OFF_HBFB);
  bf16_t* ctxbf  = (bf16_t*)(w + OFF_CTXBF);
  bf16_t* xbf    = (bf16_t*)(w + OFF_XBF);
  bf16_t* hbfall = (bf16_t*)(w + OFF_HBFALL);
  float*  pmaxc  = (float*)(w + OFF_PMAXC);
  float*  psumc  = (float*)(w + OFF_PSUMC);
  float*  pmaxb  = (float*)(w + OFF_PMAXB);
  float*  psumb  = (float*)(w + OFF_PSUMB);
  float*  cetgt  = (float*)(w + OFF_CETGT);
  float*  bowtgt = (float*)(w + OFF_BOWTGT);
  unsigned* bar  = (unsigned*)(w + OFF_BAR);

  k_init<<<1, 1, 0, stream>>>(bar);
  k_mean_ctx<<<(cfg::B * cfg::C + 255) / 256, 256, 0, stream>>>(ctx, mean);
  k_h0<<<8, 256, 0, stream>>>(mean, Wc2h, bc2h, hcur, hbfA);
  k_gix<<<96 * 12, 256, 0, stream>>>(emb, Wih, bih, gix);
  k_recurrence<<<cfg::RWG, cfg::RTH, 0, stream>>>(ctx, Wattn, Wih, Whh, bhh, gix, q,
                                                  sco, alp, hcur, hbfA, hbfB, ctxbf,
                                                  hall, call_, bar);
  k_pack<<<(cfg::R * cfg::XK + 255) / 256, 256, 0, stream>>>(hall, call_, xbf, hbfall);
  k_lse_partials<cfg::XK><<<96 * cfg::LSE_NCHUNK, 256, 0, stream>>>(xbf, Wout, bout,
                                                                    pmaxc, psumc);
  k_lse_partials<cfg::H><<<96 * cfg::LSE_NCHUNK, 256, 0, stream>>>(hbfall, Wbow, bbow,
                                                                   pmaxb, psumb);
  k_tgt<<<cfg::R / 8, 256, 0, stream>>>(hall, call_, Wout, bout, Wbow, bbow, tgt,
                                        cetgt, bowtgt);
  k_finalize<<<1, 256, 0, stream>>>(pmaxc, psumc, pmaxb, psumb, cetgt, bowtgt, mask,
                                    out);
}
